// PointNetpp_89034672046902
// MI455X (gfx1250) — matrix operand order repaired and checked by reading, not compiled, not run on hardware
//
#include <hip/hip_runtime.h>
#include <hip/hip_bf16.h>
#include <cstdint>
#include <cstddef>

// ---------------------------------------------------------------------------
// PointNet++ classification forward for MI455X (gfx1250, wave32, WMMA).
//
//  * ~54 GFLOP of GEMM vs tens of MB of traffic -> fusion/latency bound.
//    Each SA stage = ONE fused kernel per sampled group: ball-query ->
//    grouping -> 3x (GEMM+BN+ReLU) -> channel max-pool, activations in LDS
//    (320 KB/WGP; SA3 working set = 271 KB fits).
//  * Matrix math: v_wmma_f32_16x16x32_f16. Weights are PRE-PACKED once to
//    f16 with K zero-padded to a multiple of 32, so the B-fragment load is
//    16 contiguous unguarded halves (2x global_load_b128) -- the round-0
//    version's per-element guarded loads serialized the whole inner loop.
//  * Stage dims are template parameters: loops fully unroll, and the final
//    streamed layer keeps ROWS/16 (<=8) f32 accumulators in registers so one
//    B fragment feeds up to 8 back-to-back WMMAs before the next load.
//  * FPS stays the serial bottleneck (dependent argmax chain): one block per
//    batch, register-resident distances, LDS argmax tree.
// ---------------------------------------------------------------------------

#define EPS_BN 1e-5f

typedef __attribute__((ext_vector_type(16))) _Float16 v16h;
typedef __attribute__((ext_vector_type(8)))  float    v8f;

struct BNP {
  const float* b;     // bias [Cout]
  const float* g;     // gamma
  const float* beta;
  const float* mean;
  const float* var;
};

struct SAArgs {
  const float* xyz;      // [B, N, 3]
  const float* pts;      // [B, N, Cp]
  const float* new_xyz;  // [B, S, 3] (mode 0)
  float*       out;      // [B, S, C3]
  const _Float16* Wh0;   // packed f16 [C1, C0P]
  const _Float16* Wh1;   // packed f16 [C2, C1]
  const _Float16* Wh2;   // packed f16 [C3, C2]
  BNP bn0, bn1, bn2;
  int N, S, Cp;
  float radius2;
};

// f32 [Cout, Cin] -> f16 [Cout, kpad] with zero K-padding.
__global__ void pack_weights_kernel(const float* __restrict__ W,
                                    _Float16* __restrict__ Wh,
                                    int Cout, int Cin, int kpad)
{
  const int i = blockIdx.x * blockDim.x + threadIdx.x;
  if (i >= Cout * kpad) return;
  const int co = i / kpad, k = i % kpad;
  Wh[i] = (k < Cin) ? (_Float16)W[(size_t)co * Cin + k] : (_Float16)0.f;
}

// ---------------------------------------------------------------------------
// WMMA fragment loaders.
// A (16-bit, 16x32): lanes 0-15 row=lane hold K0..15; lanes 16-31
//   row=lane-16 hold K16..31 (same k order as B).
// B (16-bit, 32x16): lanes 0-15 col=lane hold K0..15; lanes 16-31 col=lane-16
//   hold K16..31 (same k order as A).
// ---------------------------------------------------------------------------
template<int LD>
__device__ __forceinline__ v16h load_a_frag(const _Float16* __restrict__ actIn,
                                            int row, int k0, int hi)
{
  v16h a;
  const _Float16* p = actIn + row * LD + k0 + (hi << 4);
#pragma unroll
  for (int j = 0; j < 8; ++j) {
    a[j]     = p[j];
    a[8 + j] = p[j + 8];
  }
  return a;
}

template<int LD>
__device__ __forceinline__ v16h load_b_frag(const _Float16* __restrict__ Wh,
                                            int col, int k0, int hi)
{
  v16h bf;
  const _Float16* p = Wh + (size_t)col * LD + k0 + (hi << 4);
#pragma unroll
  for (int h = 0; h < 16; ++h) bf[h] = p[h];   // contiguous -> 2x b128 loads
  return bf;
}

// ---------------------------------------------------------------------------
// LDS -> LDS GEMM with fused BN+ReLU epilogue (f32 accum, f16 out).
// ---------------------------------------------------------------------------
template<int ROWS, int KPAD, int COUT>
__device__ __forceinline__ void gemm_bn_relu_lds(
    const _Float16* __restrict__ actIn,
    const _Float16* __restrict__ Wh, const BNP& P,
    _Float16* __restrict__ actOut)
{
  const int lane   = threadIdx.x & 31;
  const int wave   = (int)(threadIdx.x >> 5);
  const int nwaves = (int)(blockDim.x >> 5);
  const int r15    = lane & 15;
  const int hi     = lane >> 4;
  constexpr int TM = ROWS / 16;
  constexpr int TN = COUT / 16;

  for (int t = wave; t < TM * TN; t += nwaves) {
    const int mt  = t % TM;
    const int nt  = t / TM;
    const int row = (mt << 4) + r15;
    const int col = (nt << 4) + r15;
    v8f acc = {0.f, 0.f, 0.f, 0.f, 0.f, 0.f, 0.f, 0.f};
    __builtin_prefetch(&Wh[(size_t)col * KPAD], 0, 0);
#pragma unroll
    for (int k0 = 0; k0 < KPAD; k0 += 32) {
      const v16h a  = load_a_frag<KPAD>(actIn, row, k0, hi);
      const v16h bf = load_b_frag<KPAD>(Wh, col, k0, hi);
      acc = __builtin_amdgcn_wmma_f32_16x16x32_f16(
          false, a, false, bf, (short)0, acc, false, false);
    }
    const float bb = P.b[col];
    const float sc = rsqrtf(P.var[col] + EPS_BN) * P.g[col];
    const float mm = P.mean[col];
    const float bt = P.beta[col];
#pragma unroll
    for (int rr = 0; rr < 8; ++rr) {
      const int m = (mt << 4) + (hi << 3) + rr;   // D layout: lanes16-31 M+=8
      float y = (acc[rr] + bb - mm) * sc + bt;
      actOut[m * COUT + col] = (_Float16)fmaxf(y, 0.f);
    }
  }
  __syncthreads();
}

// ---------------------------------------------------------------------------
// Final layer streamed into the channel max-pool: per column tile, all M-tile
// accumulators live in registers; one B fragment feeds TM back-to-back WMMAs.
// ---------------------------------------------------------------------------
template<int ROWS, int KPAD, int COUT>
__device__ __forceinline__ void gemm_bn_relu_maxpool(
    const _Float16* __restrict__ actIn,
    const _Float16* __restrict__ Wh, const BNP& P,
    float* __restrict__ outPooled)
{
  const int lane   = threadIdx.x & 31;
  const int wave   = (int)(threadIdx.x >> 5);
  const int nwaves = (int)(blockDim.x >> 5);
  const int r15    = lane & 15;
  const int hi     = lane >> 4;
  constexpr int TM = ROWS / 16;
  constexpr int TN = COUT / 16;

  for (int nt = wave; nt < TN; nt += nwaves) {
    const int col = (nt << 4) + r15;
    v8f acc[TM];
#pragma unroll
    for (int mt = 0; mt < TM; ++mt)
      acc[mt] = (v8f){0.f, 0.f, 0.f, 0.f, 0.f, 0.f, 0.f, 0.f};
    __builtin_prefetch(&Wh[(size_t)col * KPAD], 0, 0);
#pragma unroll
    for (int k0 = 0; k0 < KPAD; k0 += 32) {
      const v16h bf = load_b_frag<KPAD>(Wh, col, k0, hi);
#pragma unroll
      for (int mt = 0; mt < TM; ++mt) {
        const v16h a = load_a_frag<KPAD>(actIn, (mt << 4) + r15, k0, hi);
        acc[mt] = __builtin_amdgcn_wmma_f32_16x16x32_f16(
            false, a, false, bf, (short)0, acc[mt], false, false);
      }
    }
    const float bb = P.b[col];
    const float sc = rsqrtf(P.var[col] + EPS_BN) * P.g[col];
    const float mm = P.mean[col];
    const float bt = P.beta[col];
    float cmax = 0.f;   // ReLU outputs >= 0, so 0 is the pooling identity
#pragma unroll
    for (int mt = 0; mt < TM; ++mt)
#pragma unroll
      for (int rr = 0; rr < 8; ++rr) {
        const float y = (acc[mt][rr] + bb - mm) * sc + bt;
        cmax = fmaxf(cmax, fmaxf(y, 0.f));
      }
    cmax = fmaxf(cmax, __shfl_xor(cmax, 16, 32));  // combine lane halves
    if (hi == 0) outPooled[col] = cmax;
  }
}

// ---------------------------------------------------------------------------
// Fused SA stage: ball-query -> group -> MLP(3 WMMA layers) -> max-pool.
// One block = one (batch, sampled-point) group. 256 threads = 8 waves.
// ---------------------------------------------------------------------------
template<int NS, int C0, int C0P, int C1, int C2, int C3, int MODE>
__global__ void __launch_bounds__(256) sa_fused_kernel(SAArgs args)
{
  extern __shared__ __align__(16) char smem[];
  int*      sIdx = (int*)smem;                   // [<=128]
  _Float16* act0 = (_Float16*)(smem + 528);      // [NS, C0P]
  _Float16* act1 = act0 + NS * C0P;              // [NS, C1]
  _Float16* act2 = act1 + NS * C1;               // [NS, C2]

  const int b   = blockIdx.x / args.S;
  const int s   = blockIdx.x % args.S;
  const int tid = threadIdx.x;

  // --- 1. group index selection -------------------------------------------
  if (MODE == 0) {
    // Ball query: first NS indices (ascending) with sqdist <= r^2, padded
    // with the first hit. Wave 0 only, wave32 ballot + popcount prefix.
    if (tid < 32) {
      const int lane = tid;
      const float* ctr = args.new_xyz + ((size_t)b * args.S + s) * 3;
      const float cx = ctr[0], cy = ctr[1], cz = ctr[2];
      int cnt = 0;
      for (int base = 0; base < args.N && cnt < NS; base += 32) {
        const int i = base + lane;
        bool in = false;
        if (i < args.N) {
          const float* q = args.xyz + ((size_t)b * args.N + i) * 3;
          const float dx = q[0] - cx, dy = q[1] - cy, dz = q[2] - cz;
          in = (dx * dx + dy * dy + dz * dz) <= args.radius2;
        }
        const unsigned m32 = (unsigned)__ballot(in);
        const int pos = cnt + __popc(m32 & ((1u << lane) - 1u));
        if (in && pos < NS) sIdx[pos] = i;
        cnt += __popc(m32);
      }
      if (cnt < NS) {
        const int first = sIdx[0];
        for (int pp = cnt + lane; pp < NS; pp += 32) sIdx[pp] = first;
      }
    }
  } else {
    for (int r = tid; r < NS; r += blockDim.x) sIdx[r] = r;
  }
  __syncthreads();

  // --- 2. grouped features in LDS (f16, K zero-padded to C0P) -------------
  {
    const float* ctr = (MODE == 0)
        ? (args.new_xyz + ((size_t)b * args.S + s) * 3) : nullptr;
    for (int t = tid; t < NS * C0P; t += blockDim.x) {
      const int r = t / C0P, c = t % C0P;
      const int src = sIdx[r];
      float v = 0.f;
      if (c < 3) {
        v = args.xyz[((size_t)b * args.N + src) * 3 + c];
        if (MODE == 0) v -= ctr[c];
      } else if (c < C0) {
        v = args.pts[((size_t)b * args.N + src) * args.Cp + (c - 3)];
      }
      act0[t] = (_Float16)v;
    }
  }
  __syncthreads();

  // --- 3. MLP -------------------------------------------------------------
  gemm_bn_relu_lds<NS, C0P, C1>(act0, args.Wh0, args.bn0, act1);
  gemm_bn_relu_lds<NS, C1,  C2>(act1, args.Wh1, args.bn1, act2);
  gemm_bn_relu_maxpool<NS, C2, C3>(act2, args.Wh2, args.bn2,
      args.out + ((size_t)b * args.S + s) * C3);
}

// ---------------------------------------------------------------------------
// Farthest point sampling: one block per batch; register-resident distances
// (N<=4096, 256 threads -> 16/thread); LDS argmax tree, low-index tie-break.
// ---------------------------------------------------------------------------
__global__ void __launch_bounds__(256) fps_kernel(
    const float* __restrict__ xyz, int* __restrict__ idx, int N, int npoint)
{
  const int b   = blockIdx.x;
  const int tid = threadIdx.x;
  const int T   = blockDim.x;
  const float* p = xyz + (size_t)b * N * 3;
  const int cnt = (N + T - 1) / T;           // <= 16

  float dist[16];
  for (int j = 0; j < 16; ++j) dist[j] = 1e10f;

  __shared__ float sVal[256];
  __shared__ int   sInd[256];

  int far = 0;
  for (int it = 0; it < npoint; ++it) {
    if (tid == 0) idx[(size_t)b * npoint + it] = far;
    const float fx = p[far * 3 + 0], fy = p[far * 3 + 1], fz = p[far * 3 + 2];
    float bestV = -1.f;
    int   bestI = N;
    for (int j = 0; j < cnt; ++j) {
      const int i = tid + j * T;
      if (i < N) {
        const float dx = p[i * 3] - fx, dy = p[i * 3 + 1] - fy, dz = p[i * 3 + 2] - fz;
        const float nd = fminf(dist[j], dx * dx + dy * dy + dz * dz);
        dist[j] = nd;
        if (nd > bestV) { bestV = nd; bestI = i; }
      }
    }
    sVal[tid] = bestV; sInd[tid] = bestI;
    __syncthreads();
    for (int st = T >> 1; st > 0; st >>= 1) {
      if (tid < st) {
        const float ov = sVal[tid + st]; const int oi = sInd[tid + st];
        if (ov > sVal[tid] || (ov == sVal[tid] && oi < sInd[tid])) {
          sVal[tid] = ov; sInd[tid] = oi;
        }
      }
      __syncthreads();
    }
    far = sInd[0];
    __syncthreads();
  }
}

// x [B,6,N] -> xyz [B,N,3], nrm [B,N,3]
__global__ void split_transpose_kernel(const float* __restrict__ x,
                                       float* __restrict__ xyz,
                                       float* __restrict__ nrm, int B, int N)
{
  const long long i = (long long)blockIdx.x * blockDim.x + threadIdx.x;
  if (i >= (long long)B * N) return;
  const int b = (int)(i / N), n = (int)(i % N);
  const float* xb = x + (size_t)b * 6 * N;
  float* o1 = xyz + (size_t)i * 3;
  float* o2 = nrm + (size_t)i * 3;
#pragma unroll
  for (int c = 0; c < 3; ++c) {
    o1[c] = xb[(size_t)c * N + n];
    o2[c] = xb[(size_t)(3 + c) * N + n];
  }
}

// out[b,s,:] = xyz[b, idx[b,s], :]
__global__ void gather_points_kernel(const float* __restrict__ xyz,
                                     const int* __restrict__ idx,
                                     float* __restrict__ out, int N, int S, int B)
{
  const int i = blockIdx.x * blockDim.x + threadIdx.x;
  if (i >= B * S) return;
  const int b = i / S, s = i % S;
  const int src = idx[(size_t)b * S + s];
  const float* p = xyz + ((size_t)b * N + src) * 3;
  float* o = out + (size_t)i * 3;
  o[0] = p[0]; o[1] = p[1]; o[2] = p[2];
}

// FC head: one block per batch element (tiny FLOPs, plain VALU is fine).
__global__ void __launch_bounds__(256) fc_head_kernel(
    const float* __restrict__ l3,
    const float* fc1W, const float* fc1b,
    const float* bn1g, const float* bn1b, const float* bn1m, const float* bn1v,
    const float* fc2W, const float* fc2b,
    const float* bn2g, const float* bn2b, const float* bn2m, const float* bn2v,
    const float* fc3W, const float* fc3b,
    float* __restrict__ logits)
{
  __shared__ float h1[512];
  __shared__ float h2[256];
  const int b = blockIdx.x;
  const int tid = threadIdx.x;
  const float* xrow = l3 + (size_t)b * 1024;
  for (int o = tid; o < 512; o += blockDim.x) {
    float s = fc1b[o];
    const float* w = fc1W + (size_t)o * 1024;
    for (int k = 0; k < 1024; ++k) s += xrow[k] * w[k];
    s = (s - bn1m[o]) * rsqrtf(bn1v[o] + EPS_BN) * bn1g[o] + bn1b[o];
    h1[o] = fmaxf(s, 0.f);
  }
  __syncthreads();
  for (int o = tid; o < 256; o += blockDim.x) {
    float s = fc2b[o];
    const float* w = fc2W + (size_t)o * 512;
    for (int k = 0; k < 512; ++k) s += h1[k] * w[k];
    s = (s - bn2m[o]) * rsqrtf(bn2v[o] + EPS_BN) * bn2g[o] + bn2b[o];
    h2[o] = fmaxf(s, 0.f);
  }
  __syncthreads();
  if (tid < 10) {
    float s = fc3b[tid];
    const float* w = fc3W + (size_t)tid * 256;
    for (int k = 0; k < 256; ++k) s += h2[k] * w[k];
    logits[(size_t)b * 10 + tid] = s;
  }
}

// ---------------------------------------------------------------------------
// Host side.
// Assumed d_in flattening (setup_inputs insertion order, leaves in order):
//   [0] x [32,6,4096]
//   sa1 @ base 1 : W0,W1,W2, b0..2, g0..2, beta0..2, mean0..2, var0..2 (18)
//   sa2 @ base 19, sa3 @ base 37
//   [55] fc1W [56] fc1b [57..60] bn1(g,beta,mean,var)
//   [61] fc2W [62] fc2b [63..66] bn2(g,beta,mean,var)
//   [67] fc3W [68] fc3b
// Output: logits [32,10] then l3 [32,1024], both f32.
// ---------------------------------------------------------------------------
static inline size_t sa_shmem_bytes(int ns, int c0p, int c1, int c2) {
  return (size_t)528 + 2ull * (size_t)ns * (size_t)(c0p + c1 + c2);
}

extern "C" void kernel_launch(void* const* d_in, const int* in_sizes, int n_in,
                              void* d_out, int out_size, void* d_ws, size_t ws_size,
                              hipStream_t stream)
{
  (void)in_sizes; (void)n_in; (void)out_size; (void)ws_size;
  const int B = 32, N = 4096;
  const float* x = (const float*)d_in[0];
  auto F = [&](int i) { return (const float*)d_in[i]; };
  auto mkBN = [&](int base, int l) {
    BNP p;
    p.b    = F(base + 3 + l);
    p.g    = F(base + 6 + l);
    p.beta = F(base + 9 + l);
    p.mean = F(base + 12 + l);
    p.var  = F(base + 15 + l);
    return p;
  };

  // Workspace partition (~18 MB total).
  char* wsp = (char*)d_ws;
  auto alloc = [&](size_t bytes) {
    void* p = (void*)wsp;
    wsp += (bytes + 255) & ~(size_t)255;
    return p;
  };
  float* xyz = (float*)alloc((size_t)B * N * 3 * 4);
  float* nrm = (float*)alloc((size_t)B * N * 3 * 4);
  int*   fi1 = (int*)  alloc((size_t)B * 512 * 4);
  float* nx1 = (float*)alloc((size_t)B * 512 * 3 * 4);
  float* l1p = (float*)alloc((size_t)B * 512 * 128 * 4);
  int*   fi2 = (int*)  alloc((size_t)B * 128 * 4);
  float* nx2 = (float*)alloc((size_t)B * 128 * 3 * 4);
  float* l2p = (float*)alloc((size_t)B * 128 * 256 * 4);
  // Packed f16 weights [Cout, kpad].
  _Float16* w1p[3] = { (_Float16*)alloc(64u * 32 * 2),
                       (_Float16*)alloc(64u * 64 * 2),
                       (_Float16*)alloc(128u * 64 * 2) };
  _Float16* w2p[3] = { (_Float16*)alloc(128u * 160 * 2),
                       (_Float16*)alloc(128u * 128 * 2),
                       (_Float16*)alloc(256u * 128 * 2) };
  _Float16* w3p[3] = { (_Float16*)alloc(256u * 288 * 2),
                       (_Float16*)alloc(512u * 256 * 2),
                       (_Float16*)alloc(1024u * 512 * 2) };

  // --- pack weights (f32 -> f16, K zero-padded) ---------------------------
  auto pack = [&](const float* W, _Float16* Wh, int Cout, int Cin, int kpad) {
    const int tot = Cout * kpad;
    pack_weights_kernel<<<(tot + 255) / 256, 256, 0, stream>>>(W, Wh, Cout, Cin, kpad);
  };
  pack(F(1), w1p[0],  64,   6,  32); pack(F(2), w1p[1],  64,  64,  64); pack(F(3), w1p[2], 128,  64,  64);
  pack(F(19), w2p[0], 128, 131, 160); pack(F(20), w2p[1], 128, 128, 128); pack(F(21), w2p[2], 256, 128, 128);
  pack(F(37), w3p[0], 256, 259, 288); pack(F(38), w3p[1], 512, 256, 256); pack(F(39), w3p[2], 1024, 512, 512);

  // SA3 instantiation needs 271 KB of dynamic LDS (<= 320 KB/WGP on CDNA5).
  (void)hipFuncSetAttribute(
      (const void*)sa_fused_kernel<128, 259, 288, 256, 512, 1024, 1>,
      hipFuncAttributeMaxDynamicSharedMemorySize, 300 * 1024);

  // --- input split / transpose --------------------------------------------
  split_transpose_kernel<<<(B * N + 255) / 256, 256, 0, stream>>>(x, xyz, nrm, B, N);

  // --- SA1: N=4096 -> S=512, r=0.2, ns=32, 6->64->64->128 -----------------
  fps_kernel<<<B, 256, 0, stream>>>(xyz, fi1, N, 512);
  gather_points_kernel<<<(B * 512 + 255) / 256, 256, 0, stream>>>(xyz, fi1, nx1, N, 512, B);
  {
    SAArgs a{};
    a.xyz = xyz; a.pts = nrm; a.new_xyz = nx1; a.out = l1p;
    a.Wh0 = w1p[0]; a.Wh1 = w1p[1]; a.Wh2 = w1p[2];
    a.bn0 = mkBN(1, 0); a.bn1 = mkBN(1, 1); a.bn2 = mkBN(1, 2);
    a.N = N; a.S = 512; a.Cp = 3; a.radius2 = 0.2f * 0.2f;
    sa_fused_kernel<32, 6, 32, 64, 64, 128, 0>
        <<<B * 512, 256, sa_shmem_bytes(32, 32, 64, 64), stream>>>(a);
  }

  // --- SA2: N=512 -> S=128, r=0.4, ns=64, 131->128->128->256 --------------
  fps_kernel<<<B, 256, 0, stream>>>(nx1, fi2, 512, 128);
  gather_points_kernel<<<(B * 128 + 255) / 256, 256, 0, stream>>>(nx1, fi2, nx2, 512, 128, B);
  {
    SAArgs a{};
    a.xyz = nx1; a.pts = l1p; a.new_xyz = nx2; a.out = l2p;
    a.Wh0 = w2p[0]; a.Wh1 = w2p[1]; a.Wh2 = w2p[2];
    a.bn0 = mkBN(19, 0); a.bn1 = mkBN(19, 1); a.bn2 = mkBN(19, 2);
    a.N = 512; a.S = 128; a.Cp = 128; a.radius2 = 0.4f * 0.4f;
    sa_fused_kernel<64, 131, 160, 128, 128, 256, 0>
        <<<B * 128, 256, sa_shmem_bytes(64, 160, 128, 128), stream>>>(a);
  }

  // --- SA3 (global): 128 rows, 259->256->512->1024, pool to l3 ------------
  float* out = (float*)d_out;
  float* l3  = out + (size_t)B * 10;   // second tuple output region
  {
    SAArgs a{};
    a.xyz = nx2; a.pts = l2p; a.new_xyz = nullptr; a.out = l3;
    a.Wh0 = w3p[0]; a.Wh1 = w3p[1]; a.Wh2 = w3p[2];
    a.bn0 = mkBN(37, 0); a.bn1 = mkBN(37, 1); a.bn2 = mkBN(37, 2);
    a.N = 128; a.S = 1; a.Cp = 256; a.radius2 = 0.f;
    sa_fused_kernel<128, 259, 288, 256, 512, 1024, 1>
        <<<B, 256, sa_shmem_bytes(128, 288, 256, 512), stream>>>(a);
  }

  // --- FC head -> logits ---------------------------------------------------
  fc_head_kernel<<<B, 256, 0, stream>>>(
      l3,
      F(55), F(56), F(57), F(58), F(59), F(60),
      F(61), F(62), F(63), F(64), F(65), F(66),
      F(67), F(68),
      out);
}
